// CrossAttention_58171037057444
// MI455X (gfx1250) — compile-verified, matrix operand order checked by reading, not run on hardware
//
#include <hip/hip_runtime.h>
#include <hip/hip_bf16.h>

typedef unsigned short ushort_t;
typedef __bf16 bf16_t;
typedef __attribute__((ext_vector_type(16))) bf16_t v16bf;
typedef __attribute__((ext_vector_type(8)))  float  v8f;

#define DIM   1024          // all inner dims are 1024
#define NQ    2048
#define NKV   2048
#define BATCH 8
#define MTOT  (BATCH * NQ)  // 16384 flattened rows

// ---------- helpers ----------

__device__ __forceinline__ ushort_t f2bf(float f) {
  unsigned u = __float_as_uint(f);
  u += 0x7fffu + ((u >> 16) & 1u);   // round-to-nearest-even
  return (ushort_t)(u >> 16);
}

union FragAB {
  v16bf bf;
  uint4 q[2];
};

// Load a 16x32 bf16 fragment (A layout; B uses the same layout with
// lane = output column and the source matrix stored [N][K]).
// ISA 16-bit A 16x32: lanes 0-15 row M=lane hold K=0..7 (v0-3), K=16..23 (v4-7);
// lanes 16-31 hold K=8..15 and K=24..31 of the same rows.
__device__ __forceinline__ void load_frag(FragAB& f, const ushort_t* base,
                                          int ld, int lane) {
  const int l  = lane & 15;
  const int hi = lane >> 4;
  const ushort_t* p = base + (long)l * ld + hi * 8;
  f.q[0] = *(const uint4*)(p);
  f.q[1] = *(const uint4*)(p + 16);
}

__device__ __forceinline__ v8f wmma_bf16(const FragAB& a, const FragAB& b, v8f c) {
  return __builtin_amdgcn_wmma_f32_16x16x32_bf16(
      false, a.bf, false, b.bf, (short)0, c, false, false);
}

// ---------- weight transpose + convert: Wt[n][k] = bf16(W[k][n]) ----------

__global__ __launch_bounds__(256) void wtrans_kernel(const float* __restrict__ W,
                                                     ushort_t* __restrict__ Wt) {
  int t = blockIdx.x * 256 + threadIdx.x;   // 0 .. 1024*1024-1
  int k = t >> 10;
  int n = t & 1023;
  Wt[n * DIM + k] = f2bf(W[t]);
}

// ---------- tiled bf16 GEMM: Out(M,N) = X(M,K) * Wt(N,K)^T ----------
// workgroup: 256 thr = 8 waves; tile 128x128; wave w -> rows 16w..16w+15,
// 8 column tiles of 16. TRANS_OUT stores Out[n*M+m] (for v_projT).

template <bool IN_F32, bool TRANS_OUT>
__global__ __launch_bounds__(256) void gemm_bf16_kernel(
    const void* __restrict__ Xv, const ushort_t* __restrict__ Wt,
    ushort_t* __restrict__ Out, int M, int N, int K) {
  __shared__ __align__(16) ushort_t Xs[128 * 32];   // 8 KB
  __shared__ __align__(16) ushort_t Ws[128 * 32];   // 8 KB

  const int tid  = threadIdx.x;
  const int lane = tid & 31;
  const int w    = tid >> 5;
  const int m0   = blockIdx.y * 128;
  const int n0   = blockIdx.x * 128;

  v8f acc[8];
  for (int t = 0; t < 8; ++t) acc[t] = v8f{};

  for (int kk = 0; kk < K; kk += 32) {
    if (IN_F32) {
      const float* X = (const float*)Xv;
      for (int j = 0; j < 4; ++j) {
        int i = (tid + 256 * j) * 4;        // 0..4095
        int row = i >> 5, col = i & 31;
        const float4 v = *(const float4*)(X + (long)(m0 + row) * K + kk + col);
        ushort_t* d = &Xs[row * 32 + col];
        d[0] = f2bf(v.x); d[1] = f2bf(v.y); d[2] = f2bf(v.z); d[3] = f2bf(v.w);
      }
    } else {
      const ushort_t* X = (const ushort_t*)Xv;
      for (int j = 0; j < 2; ++j) {
        int i = (tid + 256 * j) * 8;
        int row = i >> 5, col = i & 31;
        *(uint4*)&Xs[row * 32 + col] =
            *(const uint4*)(X + (long)(m0 + row) * K + kk + col);
      }
    }
    for (int j = 0; j < 2; ++j) {
      int i = (tid + 256 * j) * 8;
      int row = i >> 5, col = i & 31;
      *(uint4*)&Ws[row * 32 + col] =
          *(const uint4*)(Wt + (long)(n0 + row) * K + kk + col);
    }
    __syncthreads();

    FragAB a;
    load_frag(a, Xs + (16 * w) * 32, 32, lane);
    for (int t = 0; t < 8; ++t) {
      FragAB b;
      load_frag(b, Ws + (16 * t) * 32, 32, lane);
      acc[t] = wmma_bf16(a, b, acc[t]);
    }
    __syncthreads();
  }

  const int l = lane & 15, hi = lane >> 4;
  for (int t = 0; t < 8; ++t) {
    for (int r = 0; r < 8; ++r) {
      int m = m0 + 16 * w + r + 8 * hi;
      int n = n0 + 16 * t + l;
      if (TRANS_OUT) Out[(long)n * M + m] = f2bf(acc[t][r]);
      else           Out[(long)m * N + n] = f2bf(acc[t][r]);
    }
  }
}

// ---------- flash attention ----------
// grid (NQ/16, BATCH), 256 thr = 8 waves. Workgroup owns 16 q rows.
// Loop over KV blocks of 128: wave w computes S columns [16w,16w+16);
// online softmax in LDS; wave w accumulates output columns [128w,128w+128).

__global__ __launch_bounds__(256) void attn_kernel(
    const ushort_t* __restrict__ Qp, const ushort_t* __restrict__ Kp,
    const ushort_t* __restrict__ VpT, float* __restrict__ Out) {
  __shared__ __align__(16) ushort_t Qs[16 * DIM];  // 32 KB
  __shared__ __align__(16) float    Ss[16 * 128];  // 8 KB raw scores
  __shared__ __align__(16) ushort_t Ps[16 * 128];  // 4 KB exp(P) bf16
  __shared__ float mstate[16], lstate[16], resc[16];

  const int tid  = threadIdx.x;
  const int lane = tid & 31;
  const int w    = tid >> 5;
  const int q0   = blockIdx.x * 16;
  const int b    = blockIdx.y;
  const int l    = lane & 15, hi = lane >> 4;

  const ushort_t* Qb = Qp  + ((long)b * NQ + q0) * DIM;
  const ushort_t* Kb = Kp  + (long)b * NKV * DIM;
  const ushort_t* Vb = VpT + (long)b * NKV;   // VpT is [F][MTOT], row stride MTOT
  float*          Ob = Out + ((long)b * NQ + q0) * DIM;

  for (int j = 0; j < 8; ++j) {
    int i = (tid + 256 * j) * 8;              // 16*1024 ushorts
    *(uint4*)&Qs[i] = *(const uint4*)(Qb + i);
  }
  if (tid < 16) { mstate[tid] = -1e30f; lstate[tid] = 0.f; }
  __syncthreads();

  v8f acc[8];
  for (int t = 0; t < 8; ++t) acc[t] = v8f{};

  for (int kb = 0; kb < NKV; kb += 128) {
    // ---- S(16 x 16) for this wave's KV columns: full K=1024 sweep
    v8f s = v8f{};
    const ushort_t* kbase = Kb + (long)(kb + 16 * w) * DIM;
    for (int j = 0; j < 32; ++j) {
      FragAB a, bf;
      load_frag(a,  Qs   + 32 * j, DIM, lane);
      load_frag(bf, kbase + 32 * j, DIM, lane);
      s = wmma_bf16(a, bf, s);
    }
    for (int r = 0; r < 8; ++r)
      Ss[(r + 8 * hi) * 128 + 16 * w + l] = s[r];
    __syncthreads();

    // ---- online softmax over the 128-wide block (16 threads per row)
    {
      int rr = tid >> 4;
      int c0 = (tid & 15) * 8;
      float v[8], mloc = -1e30f;
      for (int i = 0; i < 8; ++i) {
        v[i] = Ss[rr * 128 + c0 + i] * 0.03125f;   // 1/sqrt(1024)
        mloc = fmaxf(mloc, v[i]);
      }
      for (int off = 1; off < 16; off <<= 1)
        mloc = fmaxf(mloc, __shfl_xor(mloc, off, 32));
      float mold = mstate[rr];
      float mnew = fmaxf(mold, mloc);
      float lloc = 0.f;
      for (int i = 0; i < 8; ++i) {
        float p = __expf(v[i] - mnew);
        lloc += p;
        Ps[rr * 128 + c0 + i] = f2bf(p);
      }
      for (int off = 1; off < 16; off <<= 1)
        lloc += __shfl_xor(lloc, off, 32);
      if ((tid & 15) == 0) {
        float sc = __expf(mold - mnew);
        lstate[rr] = lstate[rr] * sc + lloc;
        mstate[rr] = mnew;
        resc[rr]   = sc;
      }
    }
    __syncthreads();

    // ---- rescale accumulators, then O += P(16x128) * V(128 x 128w-slice)
    for (int r = 0; r < 8; ++r) {
      float sc = resc[r + 8 * hi];
      for (int t = 0; t < 8; ++t) acc[t][r] *= sc;
    }
    FragAB pa[4];
    for (int j = 0; j < 4; ++j) load_frag(pa[j], Ps + 32 * j, 128, lane);
    for (int t = 0; t < 8; ++t) {
      const ushort_t* vbase = Vb + (long)(128 * w + 16 * t) * MTOT + kb;
      for (int j = 0; j < 4; ++j) {
        FragAB bf;
        load_frag(bf, vbase + 32 * j, MTOT, lane);
        acc[t] = wmma_bf16(pa[j], bf, acc[t]);
      }
    }
    __syncthreads();
  }

  // ---- normalize and store fp32
  for (int r = 0; r < 8; ++r) {
    float inv = 1.f / lstate[r + 8 * hi];
    for (int t = 0; t < 8; ++t)
      Ob[(long)(r + 8 * hi) * DIM + 128 * w + 16 * t + l] = acc[t][r] * inv;
  }
}

// ---------- host ----------

extern "C" void kernel_launch(void* const* d_in, const int* in_sizes, int n_in,
                              void* d_out, int out_size, void* d_ws, size_t ws_size,
                              hipStream_t stream) {
  const float* q  = (const float*)d_in[0];
  const float* y  = (const float*)d_in[1];
  const float* Wq = (const float*)d_in[2];
  const float* Wk = (const float*)d_in[3];
  const float* Wv = (const float*)d_in[4];
  float* out = (float*)d_out;

  char* ws = (char*)d_ws;
  const size_t MB = 1u << 20;
  ushort_t* WqT = (ushort_t*)(ws + 0 * MB);   // 2 MB each
  ushort_t* WkT = (ushort_t*)(ws + 2 * MB);
  ushort_t* WvT = (ushort_t*)(ws + 4 * MB);
  ushort_t* Qp  = (ushort_t*)(ws + 6 * MB);   // 32 MB  [MTOT][DIM] bf16
  ushort_t* Kp  = (ushort_t*)(ws + 38 * MB);  // 32 MB  [MTOT][DIM] bf16
  ushort_t* VpT = (ushort_t*)(ws + 70 * MB);  // 32 MB  [DIM][MTOT] bf16 (transposed)

  // 1. weights -> bf16, transposed to [N][K]
  wtrans_kernel<<<4096, 256, 0, stream>>>(Wq, WqT);
  wtrans_kernel<<<4096, 256, 0, stream>>>(Wk, WkT);
  wtrans_kernel<<<4096, 256, 0, stream>>>(Wv, WvT);

  // 2. projections (M = 16384, N = K = 1024)
  dim3 ggrid(DIM / 128, MTOT / 128);
  gemm_bf16_kernel<true,  false><<<ggrid, 256, 0, stream>>>(q,  WqT, Qp,  MTOT, DIM, DIM);
  gemm_bf16_kernel<true,  false><<<ggrid, 256, 0, stream>>>(y,  WkT, Kp,  MTOT, DIM, DIM);
  gemm_bf16_kernel<false, true ><<<ggrid, 256, 0, stream>>>(Kp, WvT, VpT, MTOT, DIM, DIM);

  // 3. flash attention
  attn_kernel<<<dim3(NQ / 16, BATCH), 256, 0, stream>>>(Qp, Kp, VpT, out);
}
